// ScaledDotProductAttention_3607772529050
// MI455X (gfx1250) — matrix operand order repaired and checked by reading, not compiled, not run on hardware
//
#include <hip/hip_runtime.h>

// ---------------------------------------------------------------------------
// Flash-attention, B=4, S=4096, D=128, fp32 in/out, bf16 WMMA math.
// Pass 1 (if workspace available): convert K -> bf16 [B,S,D], V -> bf16
// transposed [B,D,S] into d_ws.  Pass 2: flash attention, 64-key chunks,
// double-buffered LDS staged with async global->LDS DMA (ASYNCcnt).
// One wave = 16 query rows; 8 waves/block = 128 rows; grid = (S/128, B).
// Softmax denominator computed as a 9th output column (P x ones) via WMMA.
// ---------------------------------------------------------------------------

#if defined(__has_builtin)
#  if __has_builtin(__builtin_amdgcn_global_load_async_to_lds_b128)
#    define HAVE_ASYNC_LDS 1
#  endif
#  if __has_builtin(__builtin_amdgcn_s_wait_asynccnt)
#    define HAVE_WAIT_ASYNC 1
#  endif
#endif

typedef __attribute__((ext_vector_type(16))) __bf16 v16bf;
typedef __attribute__((ext_vector_type(8)))  float  v8f;
typedef int v4i32 __attribute__((vector_size(16)));

union FragAB { v16bf v; uint4 q[2]; unsigned short s[16]; };
union FragC  { v8f  v; float f[8]; };

__device__ __forceinline__ unsigned int f2bf(float x) {
  unsigned int u = __float_as_uint(x);
  return (u + 0x7FFFu + ((u >> 16) & 1u)) >> 16;   // RNE
}
__device__ __forceinline__ unsigned int pk2(float lo, float hi) {
  return f2bf(lo) | (f2bf(hi) << 16);
}

#if defined(HAVE_ASYNC_LDS)
typedef __attribute__((address_space(1))) v4i32* gptr_b128;
typedef __attribute__((address_space(3))) v4i32* lptr_b128;
__device__ __forceinline__ void async_cp16(const unsigned short* g, unsigned short* l) {
  __builtin_amdgcn_global_load_async_to_lds_b128(
      (gptr_b128)(v4i32*)g, (lptr_b128)(v4i32*)l, 0, 0);
}
#endif

__device__ __forceinline__ void wait_async_zero() {
#if defined(HAVE_ASYNC_LDS)
#  if defined(HAVE_WAIT_ASYNC)
  __builtin_amdgcn_s_wait_asynccnt(0);
#  else
  asm volatile("s_wait_asynccnt 0x0" ::: "memory");
#  endif
#endif
}

// ---------------------------------------------------------------------------
// Pass-1 kernels
// ---------------------------------------------------------------------------
__global__ void kconv_kernel(const float* __restrict__ src,
                             unsigned short* __restrict__ dst, long n) {
  long i = ((long)blockIdx.x * blockDim.x + threadIdx.x) * 8;
  if (i + 8 <= n) {
    float4 a = ((const float4*)(src + i))[0];
    float4 b = ((const float4*)(src + i))[1];
    uint4 o;
    o.x = pk2(a.x, a.y); o.y = pk2(a.z, a.w);
    o.z = pk2(b.x, b.y); o.w = pk2(b.z, b.w);
    *(uint4*)(dst + i) = o;
  }
}

// V [B,S,128] f32 -> Vt [B,128,S] bf16, 32x32 LDS tiles.
__global__ void vtrans_kernel(const float* __restrict__ V,
                              unsigned short* __restrict__ Vt, int S) {
  __shared__ float t[32][33];
  const int b = blockIdx.z, s0 = blockIdx.x * 32, d0 = blockIdx.y * 32;
  const int tid = threadIdx.x;
  const float* Vb = V + (size_t)b * S * 128;
  {
    int sl = tid >> 5, c = tid & 31;          // 8 rows per pass
#pragma unroll
    for (int p = 0; p < 4; ++p) {
      int s = sl + p * 8;
      t[s][c] = Vb[(size_t)(s0 + s) * 128 + d0 + c];
    }
  }
  __syncthreads();
  unsigned short* Ob = Vt + (size_t)b * 128 * S;
  {
    int dl = tid >> 4, sp = tid & 15;         // 16 d-rows per pass, pack 2 s
#pragma unroll
    for (int p = 0; p < 2; ++p) {
      int d = dl + p * 16;
      unsigned int pk = pk2(t[2 * sp][d], t[2 * sp + 1][d]);
      *(unsigned int*)&Ob[(size_t)(d0 + d) * S + s0 + 2 * sp] = pk;
    }
  }
}

// ---------------------------------------------------------------------------
// Main flash-attention kernel.  PRE=true: bf16 K/Vt from workspace (async DMA
// staging).  PRE=false: f32 K/V from inputs, convert+transpose while staging.
// ---------------------------------------------------------------------------
template <bool PRE>
__launch_bounds__(256, 1)
__global__ void fa_fwd_kernel(const float* __restrict__ Q,
                              const float* __restrict__ Kf,
                              const float* __restrict__ Vf,
                              const unsigned short* __restrict__ Kw,
                              const unsigned short* __restrict__ Vt,
                              float* __restrict__ O, int S)
{
  constexpr int D = 128;
  // 1/sqrt(D) * log2(e), folded into Q at fragment build.
  constexpr float SCALE2 = 0.08838834764831845f * 1.44269504088896340f;

  __shared__ __align__(16) unsigned short lKb[2][64 * D];   // [key][d]  bf16
  __shared__ __align__(16) unsigned short lVb[2][D * 64];   // [d][key]  bf16
  __shared__ __align__(16) unsigned short lP[8][16 * 64];   // per-wave P scratch

  const int tid  = threadIdx.x;
  const int wid  = tid >> 5;
  const int lane = tid & 31;
  const int half = lane >> 4;
  const int mrow = lane & 15;
  const int b    = blockIdx.y;
  const int qbase = blockIdx.x * 128 + wid * 16;

  const float*          Qb  = Q  + (size_t)b * S * D;
  const float*          Kfb = PRE ? nullptr : Kf + (size_t)b * S * D;
  const float*          Vfb = PRE ? nullptr : Vf + (size_t)b * S * D;
  const unsigned short* Kwb = PRE ? Kw + (size_t)b * S * D : nullptr;
  const unsigned short* Vtb = PRE ? Vt + (size_t)b * D * S : nullptr;

  // ---- staging helper: fill buffer `p` with keys [n0, n0+64) -------------
  auto stage = [&](int p, int n0) {
    unsigned short* lKd = &lKb[p][0];
    unsigned short* lVd = &lVb[p][0];
    if constexpr (PRE) {
      // K chunk: 16KB contiguous
      const unsigned short* g = Kwb + (size_t)n0 * D + tid * 32;
      unsigned short* d = lKd + tid * 32;
      // V chunk: 128 rows of 64 bf16 (two 64B halves per row)
      int dr = tid >> 1, so = (tid & 1) * 32;
      const unsigned short* gv = Vtb + (size_t)dr * S + n0 + so;
      unsigned short* dv = lVd + dr * 64 + so;
#if defined(HAVE_ASYNC_LDS)
#pragma unroll
      for (int j = 0; j < 4; ++j) async_cp16(g + j * 8, d + j * 8);
#pragma unroll
      for (int j = 0; j < 4; ++j) async_cp16(gv + j * 8, dv + j * 8);
#else
#pragma unroll
      for (int j = 0; j < 4; ++j) ((uint4*)d)[j] = ((const uint4*)g)[j];
#pragma unroll
      for (int j = 0; j < 4; ++j) ((uint4*)dv)[j] = ((const uint4*)gv)[j];
#endif
    } else {
      // K: thread handles 32 contiguous f32, convert -> bf16
      const float* ks = Kfb + (size_t)n0 * D + tid * 32;
      unsigned short* d = lKd + tid * 32;
#pragma unroll
      for (int j = 0; j < 4; ++j) {
        float4 x = ((const float4*)ks)[2 * j];
        float4 y = ((const float4*)ks)[2 * j + 1];
        uint4 o;
        o.x = pk2(x.x, x.y); o.y = pk2(x.z, x.w);
        o.z = pk2(y.x, y.y); o.w = pk2(y.z, y.w);
        ((uint4*)d)[j] = o;
      }
      // V: transpose while converting
      const float* vs = Vfb + (size_t)n0 * D + tid * 32;
      int key = tid >> 2, dbase = (tid & 3) * 32;
#pragma unroll
      for (int j = 0; j < 32; ++j)
        lVd[(dbase + j) * 64 + key] = (unsigned short)f2bf(vs[j]);
    }
  };

  // ---- Q tile (scaled) -> 4 bf16 A-fragments -----------------------------
  // A-layout per lane: row M = lane%16; elements = one contiguous 16-float
  // run at d = 32f + 16*half.
  FragAB qa[4];
  {
    const float* qr = Qb + (size_t)(qbase + mrow) * D;
#pragma unroll
    for (int f = 0; f < 4; ++f) {
      const float* r0 = qr + 32 * f + 16 * half;
      const float* r1 = qr + 32 * f + 16 * half + 8;
      float4 x0 = ((const float4*)r0)[0], x1 = ((const float4*)r0)[1];
      float4 y0 = ((const float4*)r1)[0], y1 = ((const float4*)r1)[1];
      uint4 o;
      o.x = pk2(SCALE2 * x0.x, SCALE2 * x0.y);
      o.y = pk2(SCALE2 * x0.z, SCALE2 * x0.w);
      o.z = pk2(SCALE2 * x1.x, SCALE2 * x1.y);
      o.w = pk2(SCALE2 * x1.z, SCALE2 * x1.w);
      qa[f].q[0] = o;
      o.x = pk2(SCALE2 * y0.x, SCALE2 * y0.y);
      o.y = pk2(SCALE2 * y0.z, SCALE2 * y0.w);
      o.z = pk2(SCALE2 * y1.x, SCALE2 * y1.y);
      o.w = pk2(SCALE2 * y1.z, SCALE2 * y1.w);
      qa[f].q[1] = o;
    }
  }

  // All-ones bf16 B-fragment: P x ones accumulates softmax row-sums into a
  // 9th accumulator (replicated across all lanes since every column is 1).
  FragAB ones;
#pragma unroll
  for (int i = 0; i < 16; ++i) ones.s[i] = 0x3F80;   // bf16 1.0

  FragC acc[9];            // [0..7] = O tiles, [8] = row-sum (l)
  float mstate[8];
#pragma unroll
  for (int r = 0; r < 8; ++r) {
    mstate[r] = -3.0e38f;
#pragma unroll
    for (int a = 0; a < 9; ++a) acc[a].f[r] = 0.0f;
  }

  const int nchunks = S >> 6;   // 64 keys per chunk
  stage(0, 0);

#pragma unroll 1
  for (int c = 0; c < nchunks; ++c) {
    const int p = c & 1;
    wait_async_zero();     // this wave's stage of buf[p] complete
    __syncthreads();       // all waves: buf[p] ready, buf[1-p] free

    if (c + 1 < nchunks) stage(1 - p, (c + 1) << 6);

    const unsigned short* lKd = &lKb[p][0];
    const unsigned short* lVd = &lVb[p][0];

    // ---- S tile (16x64) = (Q*scale) * K^T : 16 WMMAs ---------------------
    FragC sf[4];
#pragma unroll
    for (int t = 0; t < 4; ++t) {
      v8f a = {};
#pragma unroll
      for (int f = 0; f < 4; ++f) {
        FragAB bk;   // B-layout: K = i + 16*half, N = lane%16
        const unsigned short* src = &lKd[(16 * t + mrow) * D + 32 * f + 16 * half];
        bk.q[0] = ((const uint4*)src)[0];
        bk.q[1] = ((const uint4*)src)[1];
        a = __builtin_amdgcn_wmma_f32_16x16x32_bf16(false, qa[f].v, false, bk.v,
                                                    (short)0, a, false, false);
      }
      sf[t].v = a;
    }

    // ---- Online softmax (exp2 domain); P straight to LDS -----------------
    unsigned short* myP = &lP[wid][0];
#pragma unroll
    for (int r = 0; r < 8; ++r) {
      float a0 = sf[0].f[r], a1 = sf[1].f[r];
      float a2 = sf[2].f[r], a3 = sf[3].f[r];
      float m = fmaxf(fmaxf(a0, a1), fmaxf(a2, a3));
#pragma unroll
      for (int off = 1; off < 16; off <<= 1)
        m = fmaxf(m, __shfl_xor(m, off, 32));          // within 16-lane half
      float mnew  = fmaxf(mstate[r], m);
      float alpha = exp2f(mstate[r] - mnew);
      mstate[r]   = mnew;
      int row = r + 8 * half;
      myP[row * 64 + mrow]      = (unsigned short)f2bf(exp2f(a0 - mnew));
      myP[row * 64 + 16 + mrow] = (unsigned short)f2bf(exp2f(a1 - mnew));
      myP[row * 64 + 32 + mrow] = (unsigned short)f2bf(exp2f(a2 - mnew));
      myP[row * 64 + 48 + mrow] = (unsigned short)f2bf(exp2f(a3 - mnew));
#pragma unroll
      for (int a = 0; a < 9; ++a) acc[a].f[r] *= alpha;
    }
    asm volatile("s_wait_dscnt 0" ::: "memory");   // wave-local store->load

    // ---- P as two 16x32 A-fragments --------------------------------------
    FragAB pa0, pa1;
    {
      const unsigned short* pr = &lP[wid][mrow * 64];
      pa0.q[0] = *(const uint4*)(pr + 16 * half);
      pa0.q[1] = *(const uint4*)(pr + 16 * half + 8);
      pa1.q[0] = *(const uint4*)(pr + 32 + 16 * half);
      pa1.q[1] = *(const uint4*)(pr + 40 + 16 * half);
    }

    // ---- O += P * V (16 WMMAs); l += P * ones (2 WMMAs) ------------------
#pragma unroll
    for (int dt = 0; dt < 8; ++dt) {
      FragAB bv;
      const unsigned short* vr = &lVd[(16 * dt + mrow) * 64 + 16 * half];
      bv.q[0] = ((const uint4*)vr)[0];
      bv.q[1] = ((const uint4*)vr)[1];
      acc[dt].v = __builtin_amdgcn_wmma_f32_16x16x32_bf16(false, pa0.v, false, bv.v,
                                                          (short)0, acc[dt].v,
                                                          false, false);
      const unsigned short* vr2 = vr + 32;
      bv.q[0] = ((const uint4*)vr2)[0];
      bv.q[1] = ((const uint4*)vr2)[1];
      acc[dt].v = __builtin_amdgcn_wmma_f32_16x16x32_bf16(false, pa1.v, false, bv.v,
                                                          (short)0, acc[dt].v,
                                                          false, false);
    }
    acc[8].v = __builtin_amdgcn_wmma_f32_16x16x32_bf16(false, pa0.v, false, ones.v,
                                                       (short)0, acc[8].v,
                                                       false, false);
    acc[8].v = __builtin_amdgcn_wmma_f32_16x16x32_bf16(false, pa1.v, false, ones.v,
                                                       (short)0, acc[8].v,
                                                       false, false);
  }

  // ---- Epilogue ------------------------------------------------------------
  float linv[8];
#pragma unroll
  for (int r = 0; r < 8; ++r) linv[r] = 1.0f / acc[8].f[r];

  float* Ob = O + (size_t)b * S * D + (size_t)qbase * D;
#pragma unroll
  for (int dt = 0; dt < 8; ++dt) {
#pragma unroll
    for (int r = 0; r < 8; ++r) {
      int row = r + 8 * half;
      int d   = 16 * dt + mrow;
      Ob[(size_t)row * D + d] = acc[dt].f[r] * linv[r];
    }
  }
}

// ---------------------------------------------------------------------------
extern "C" void kernel_launch(void* const* d_in, const int* in_sizes, int n_in,
                              void* d_out, int out_size, void* d_ws, size_t ws_size,
                              hipStream_t stream) {
  (void)n_in; (void)out_size;
  const float* Q = (const float*)d_in[0];
  const float* K = (const float*)d_in[1];
  const float* V = (const float*)d_in[2];
  float* O = (float*)d_out;
  const int B = 4, D = 128;
  const int S = in_sizes[0] / (B * D);       // 4096
  const size_t elems = (size_t)B * S * D;    // per tensor
  const size_t need  = elems * 2 * 2;        // bf16 K + bf16 Vt

  dim3 grid(S / 128, B);
  if (d_ws && ws_size >= need) {
    unsigned short* Kw = (unsigned short*)d_ws;
    unsigned short* Vt = Kw + elems;
    kconv_kernel<<<dim3((unsigned)(elems / 2048)), 256, 0, stream>>>(K, Kw, (long)elems);
    vtrans_kernel<<<dim3(S / 32, D / 32, B), 256, 0, stream>>>(V, Vt, S);
    fa_fwd_kernel<true><<<grid, dim3(256), 0, stream>>>(Q, nullptr, nullptr, Kw, Vt, O, S);
  } else {
    fa_fwd_kernel<false><<<grid, dim3(256), 0, stream>>>(Q, K, V, nullptr, nullptr, O, S);
  }
}
